// FiBiNet_65867618452117
// MI455X (gfx1250) — matrix operand order repaired and checked by reading, not compiled, not run on hardware
//
#include <hip/hip_runtime.h>
#include <hip/hip_bf16.h>
#include <cmath>

typedef _Float16 half_t;
typedef __attribute__((ext_vector_type(16))) _Float16 v16h;
typedef __attribute__((ext_vector_type(8)))  _Float16 v8h;
typedef __attribute__((ext_vector_type(8)))  float    v8f;

#define B_     4096
#define NF_    26
#define V_     10000
#define D_     32
#define ND_    13
#define SF_    8
#define H1_    1024
#define H2_    512
#define NPAIR_ 325          // NF*(NF-1)/2
#define K1RAW_ 20813        // 2*NPAIR*D + ND
#define K1PAD_ 20864        // 326 * 64, zero padded

#define KC_    64           // K chunk staged in LDS per iteration
#define AST_   (KC_ + 8)    // LDS row stride (halves): 144B -> conflict-free frags
#define BST_   (KC_ + 8)

// ---------------------------------------------------------------------------
// gfx1250 async global->LDS copy (16B per lane, ASYNCcnt-tracked)
// ---------------------------------------------------------------------------
__device__ __forceinline__ void async_ld16(unsigned int lds_off, const void* g) {
  unsigned long long ga = (unsigned long long)g;
  asm volatile("global_load_async_to_lds_b128 %0, %1, off"
               : : "v"(lds_off), "v"(ga) : "memory");
}
__device__ __forceinline__ void wait_async0() {
  asm volatile("s_wait_asynccnt 0" : : : "memory");
}

// ---------------------------------------------------------------------------
// Per-row feature pipeline: gather -> SENet -> bilinear -> deep_x (f16) + linear
// One 256-thread block per batch row.
// ---------------------------------------------------------------------------
__global__ __launch_bounds__(256) void prep_kernel(
    const int*   __restrict__ sparse,        // [B][NF]
    const float* __restrict__ dense,         // [B][ND]
    const float* __restrict__ embed_table,   // [NF][V][D]
    const float* __restrict__ linear_table,  // [NF][V]
    const float* __restrict__ linear_dense_w,// [ND]
    const float* __restrict__ senet_w1,      // [NF][SF]
    const float* __restrict__ senet_w2,      // [SF][NF]
    const float* __restrict__ bilinear_w,    // [D][D]
    half_t*      __restrict__ deep_x,        // [B][K1PAD] f16
    float*       __restrict__ lin_out)       // [B]
{
  __shared__ float emb[NF_][D_];
  __shared__ float vsh[NF_][D_];
  __shared__ float bwsh[D_][D_];
  __shared__ float zsh[NF_];
  __shared__ float t1sh[SF_];
  __shared__ float Ash[NF_];
  __shared__ int   sid[NF_];
  __shared__ int   pstart[NF_ + 1];

  const int b = blockIdx.x;
  const int t = threadIdx.x;

  if (t < NF_)     sid[t]    = sparse[b * NF_ + t];
  if (t < NF_ + 1) pstart[t] = t * (2 * NF_ - t - 1) / 2;  // triu row offsets
  for (int idx = t; idx < D_ * D_; idx += 256)
    bwsh[idx >> 5][idx & 31] = bilinear_w[idx];
  __syncthreads();

  for (int idx = t; idx < NF_ * D_; idx += 256) {
    int f = idx >> 5, d = idx & 31;
    emb[f][d] = embed_table[((size_t)f * V_ + sid[f]) * D_ + d];
  }
  __syncthreads();

  if (t < NF_) {
    float s = 0.f;
#pragma unroll
    for (int d = 0; d < D_; ++d) s += emb[t][d];
    zsh[t] = s * (1.f / D_);
  }
  __syncthreads();

  if (t < SF_) {
    float s = 0.f;
#pragma unroll
    for (int f = 0; f < NF_; ++f) s += zsh[f] * senet_w1[f * SF_ + t];
    t1sh[t] = s > 0.f ? s : 0.f;
  }
  __syncthreads();

  if (t < NF_) {
    float s = 0.f;
#pragma unroll
    for (int k = 0; k < SF_; ++k) s += t1sh[k] * senet_w2[k * NF_ + t];
    Ash[t] = s > 0.f ? s : 0.f;
  }
  __syncthreads();

  // v = emb @ bilinear_w   (bilinear(senet_x) = A[i]*A[j]*bilinear(emb))
  for (int idx = t; idx < NF_ * D_; idx += 256) {
    int f = idx >> 5, e = idx & 31;
    float s = 0.f;
#pragma unroll
    for (int d = 0; d < D_; ++d) s += emb[f][d] * bwsh[d][e];
    vsh[f][e] = s;
  }
  __syncthreads();

  // deep_x row: [bi_emb (325*32) | bi_senet (325*32) | dense (13) | pad]
  half_t* dxr = deep_x + (size_t)b * K1PAD_;
  int i = 0;  // monotone scan state for pair -> (i,j)
  for (int idx = t; idx < NPAIR_ * D_; idx += 256) {
    int p = idx >> 5, d = idx & 31;
    while (pstart[i + 1] <= p) ++i;
    int j = i + 1 + (p - pstart[i]);
    float be = vsh[i][d] * vsh[j][d];
    dxr[idx]               = (half_t)be;
    dxr[NPAIR_ * D_ + idx] = (half_t)(Ash[i] * Ash[j] * be);
  }
  if (t < K1PAD_ - 2 * NPAIR_ * D_)  // 64 tail slots: 13 dense + 51 zero pad
    dxr[2 * NPAIR_ * D_ + t] = (t < ND_) ? (half_t)dense[b * ND_ + t] : (half_t)0.f;

  if (t == 0) {
    float lo = 0.f;
    for (int f = 0; f < NF_; ++f) lo += linear_table[(size_t)f * V_ + sid[f]];
    for (int d = 0; d < ND_; ++d) lo += dense[b * ND_ + d] * linear_dense_w[d];
    lin_out[b] = lo;
  }
}

// ---------------------------------------------------------------------------
// fp32 [Kin][N] -> f16 transposed [N][Kpad], zero-pad K. Coalesced both sides
// via 32x32 LDS tile. Grid: (Kpad/32, ceil(N/32)), block 256.
// ---------------------------------------------------------------------------
__global__ __launch_bounds__(256) void transpose_wt_kernel(
    const float* __restrict__ W, half_t* __restrict__ Wt,
    int Kin, int N, int Kpad)
{
  __shared__ float tile[32][33];
  const int k0 = blockIdx.x * 32, n0 = blockIdx.y * 32;
  const int tx = threadIdx.x & 31, ty = threadIdx.x >> 5;  // 32 x 8

  for (int r = ty; r < 32; r += 8) {
    int k = k0 + r, n = n0 + tx;
    tile[r][tx] = (k < Kin && n < N) ? W[(size_t)k * N + n] : 0.f;
  }
  __syncthreads();
  for (int r = ty; r < 32; r += 8) {
    int n = n0 + r, k = k0 + tx;
    if (n < N) Wt[(size_t)n * Kpad + k] = (half_t)tile[tx][r];
  }
}

// ---------------------------------------------------------------------------
// C = tanh(A @ B + bias), f16 in / f32 accumulate / f16 out.
// A row-major [M][lda], Bt transposed [N][K].
// Block: 256 threads = 8 waves (2M x 4N); block tile 64 x 256.
// Double-buffered LDS staging filled by async global->LDS copies; fragments
// read from LDS in contiguous k order (k = 16*(lane/16) + s) for A and B; 8 wmma per 32-K step.
// ---------------------------------------------------------------------------
__global__ __launch_bounds__(256) void gemm_bias_tanh_kernel(
    const half_t* __restrict__ A, const half_t* __restrict__ Bt,
    const float* __restrict__ bias, half_t* __restrict__ out,
    int N, int K, int lda)
{
  __shared__ __align__(32) half_t AshT[2][64 * AST_];    // 2 x 9KB
  __shared__ __align__(32) half_t BshT[2][256 * BST_];   // 2 x 36KB

  const int t    = threadIdx.x;
  const int lane = t & 31;
  const int wave = t >> 5;
  const int wm   = wave >> 2;            // 0..1
  const int wn   = wave & 3;             // 0..3
  const int m0b  = blockIdx.x * 64;
  const int n0b  = blockIdx.y * 256;
  const int l16  = lane & 15;
  const int koA  = (lane & 16) ? 16 : 0; // A frag: lanes 16-31 hold K=16..31
  const int koB  = (lane & 16) ? 16 : 0; // B frag: lanes 16-31 hold K=16..31

  unsigned int aLds[2] = { (unsigned int)(size_t)&AshT[0][0],
                           (unsigned int)(size_t)&AshT[1][0] };
  unsigned int bLds[2] = { (unsigned int)(size_t)&BshT[0][0],
                           (unsigned int)(size_t)&BshT[1][0] };

  // Stage one K-chunk (KC halves) of the A (64 rows) and B (256 rows) tiles.
  auto issue_chunk = [&](int kg, int nb) {
    for (int c = t; c < 64 * (KC_ / 8); c += 256) {        // 2 iters
      int row = c >> 3, ch = c & 7;
      async_ld16(aLds[nb] + (unsigned int)(row * AST_ + ch * 8) * 2,
                 A + (size_t)(m0b + row) * lda + kg + ch * 8);
    }
    for (int c = t; c < 256 * (KC_ / 8); c += 256) {       // 8 iters
      int row = c >> 3, ch = c & 7;
      async_ld16(bLds[nb] + (unsigned int)(row * BST_ + ch * 8) * 2,
                 Bt + (size_t)(n0b + row) * K + kg + ch * 8);
    }
  };

  v8f acc[2][4];
#pragma unroll
  for (int mt = 0; mt < 2; ++mt)
#pragma unroll
    for (int nt = 0; nt < 4; ++nt) {
      v8f z = {0.f, 0.f, 0.f, 0.f, 0.f, 0.f, 0.f, 0.f};
      acc[mt][nt] = z;
    }

  issue_chunk(0, 0);
  wait_async0();
  __syncthreads();

  const int nchunks = K / KC_;
  for (int ci = 0; ci < nchunks; ++ci) {
    const int cur = ci & 1;
    if (ci + 1 < nchunks) issue_chunk((ci + 1) * KC_, cur ^ 1);  // prefetch

    const half_t* Ab = &AshT[cur][0];
    const half_t* Bb = &BshT[cur][0];
#pragma unroll
    for (int kc2 = 0; kc2 < KC_; kc2 += 32) {
      v16h afr[2];
#pragma unroll
      for (int mt = 0; mt < 2; ++mt) {
        const half_t* ap = Ab + (wm * 32 + mt * 16 + l16) * AST_ + kc2;
        v8h lo = *(const v8h*)(ap + koA);
        v8h hi = *(const v8h*)(ap + 8 + koA);
        afr[mt] = __builtin_shufflevector(
            lo, hi, 0,1,2,3,4,5,6,7,8,9,10,11,12,13,14,15);
      }
      v16h bfr[4];
#pragma unroll
      for (int nt = 0; nt < 4; ++nt)
        bfr[nt] = *(const v16h*)(Bb + (wn * 64 + nt * 16 + l16) * BST_ +
                                 kc2 + koB);
#pragma unroll
      for (int mt = 0; mt < 2; ++mt)
#pragma unroll
        for (int nt = 0; nt < 4; ++nt)
          acc[mt][nt] = __builtin_amdgcn_wmma_f32_16x16x32_f16(
              false, afr[mt], false, bfr[nt], (short)0, acc[mt][nt],
              false, false);
    }

    wait_async0();     // prefetch landed (this wave's async ops)
    __syncthreads();   // all waves done reading cur / writing cur^1
  }

  // C layout: VGPR r, lanes 0-15 -> M=r, N=lane ; lanes 16-31 -> M=8+r, N=lane-16
  const int cm = (lane & 16) ? 8 : 0;
#pragma unroll
  for (int mt = 0; mt < 2; ++mt) {
#pragma unroll
    for (int nt = 0; nt < 4; ++nt) {
      int   cn = n0b + wn * 64 + nt * 16 + l16;
      float bs = bias[cn];
      int   mb = m0b + wm * 32 + mt * 16 + cm;
#pragma unroll
      for (int r = 0; r < 8; ++r)
        out[(size_t)(mb + r) * N + cn] = (half_t)tanhf(acc[mt][nt][r] + bs);
    }
  }
}

// ---------------------------------------------------------------------------
// out[b] = lin[b] + h2[b,:] . w3 + b3
// ---------------------------------------------------------------------------
__global__ __launch_bounds__(128) void final_kernel(
    const half_t* __restrict__ h2, const float* __restrict__ w3,
    const float* __restrict__ b3, const float* __restrict__ lin,
    float* __restrict__ out)
{
  __shared__ float red[128];
  const int b = blockIdx.x, t = threadIdx.x;
  float s = 0.f;
  for (int k = t; k < H2_; k += 128)
    s += (float)h2[(size_t)b * H2_ + k] * w3[k];
  red[t] = s;
  __syncthreads();
  for (int off = 64; off > 0; off >>= 1) {
    if (t < off) red[t] += red[t + off];
    __syncthreads();
  }
  if (t == 0) out[b] = lin[b] + red[0] + b3[0];
}

// ---------------------------------------------------------------------------
extern "C" void kernel_launch(void* const* d_in, const int* in_sizes, int n_in,
                              void* d_out, int out_size, void* d_ws, size_t ws_size,
                              hipStream_t stream)
{
  (void)in_sizes; (void)n_in; (void)out_size; (void)ws_size;
  const int*   sparse = (const int*)  d_in[0];
  const float* dense  = (const float*)d_in[1];
  const float* embed  = (const float*)d_in[2];
  const float* lint   = (const float*)d_in[3];
  const float* ldw    = (const float*)d_in[4];
  const float* sw1    = (const float*)d_in[5];
  const float* sw2    = (const float*)d_in[6];
  const float* bw     = (const float*)d_in[7];
  const float* w1     = (const float*)d_in[8];
  const float* bias1  = (const float*)d_in[9];
  const float* w2     = (const float*)d_in[10];
  const float* bias2  = (const float*)d_in[11];
  const float* w3     = (const float*)d_in[12];
  const float* bias3  = (const float*)d_in[13];
  float* out = (float*)d_out;

  char* p = (char*)d_ws;
  auto carve = [&](size_t bytes) {
    char* r = p;
    p += (bytes + 255) & ~(size_t)255;
    return r;
  };
  float*  lin   = (float*) carve((size_t)B_ * 4);
  half_t* deepx = (half_t*)carve((size_t)B_  * K1PAD_ * 2);   // ~171 MB
  half_t* w1t   = (half_t*)carve((size_t)H1_ * K1PAD_ * 2);   // ~43 MB
  half_t* h1    = (half_t*)carve((size_t)B_  * H1_ * 2);      // 8 MB
  half_t* w2t   = (half_t*)carve((size_t)H2_ * H1_ * 2);      // 1 MB
  half_t* h2    = (half_t*)carve((size_t)B_  * H2_ * 2);      // 4 MB

  transpose_wt_kernel<<<dim3(K1PAD_ / 32, H1_ / 32), 256, 0, stream>>>(
      w1, w1t, K1RAW_, H1_, K1PAD_);
  transpose_wt_kernel<<<dim3(H1_ / 32, H2_ / 32), 256, 0, stream>>>(
      w2, w2t, H1_, H2_, H1_);
  prep_kernel<<<B_, 256, 0, stream>>>(sparse, dense, embed, lint, ldw,
                                      sw1, sw2, bw, deepx, lin);
  gemm_bias_tanh_kernel<<<dim3(B_ / 64, H1_ / 256), 256, 0, stream>>>(
      deepx, w1t, bias1, h1, H1_, K1PAD_, K1PAD_);
  gemm_bias_tanh_kernel<<<dim3(B_ / 64, H2_ / 256), 256, 0, stream>>>(
      h1, w2t, bias2, h2, H2_, H1_, H1_);
  final_kernel<<<B_, 128, 0, stream>>>(h2, w3, bias3, lin, out);
}
